// Sampling3DOperator_80109730005014
// MI455X (gfx1250) — matrix operand order repaired and checked by reading, not compiled, not run on hardware
//
#include <hip/hip_runtime.h>
#include <hip/hip_bf16.h>

typedef __attribute__((ext_vector_type(16))) _Float16 v16h;
typedef __attribute__((ext_vector_type(8)))  float    v8f;

#define N_QUERIES  300
#define C_DIM      256
#define P_IN       32
#define N_LEVELS   4

// -----------------------------------------------------------------------------
// Kernel 1: offsets[M,3] = query_content[M,256] @ W_off[256,3] + b_off
// One wave32 per 16-row tile, 8 x v_wmma_f32_16x16x32_f16 (K=256).
// B operand staged in LDS transposed + zero-padded to [16][256] f16 so the
// per-step fragment build is two unconditional ds_load_b128 per lane.
//
// Fragment layouts used here (contiguous k, 16-bit):
//   A 16x32: lanes 0-15 row=lane,   a[0..7]=K0..7,  a[8..15]=K8..15
//            lanes 16-31 row=lane-16, a[0..7]=K16..23, a[8..15]=K24..31
//   B 32x16: lanes 0-15 col=lane,    b[j]=K j ; lanes 16-31 col=lane-16, b[j]=K16+j
//   D: VGPR r -> lanes 0-15 (M=r, N=lane); lanes 16-31 (M=8+r, N=lane-16)
// -----------------------------------------------------------------------------
__global__ void __launch_bounds__(32)
offsets_gemm_wmma_kernel(const float* __restrict__ qc,     // [M, 256]
                         const float* __restrict__ Woff,   // [256, 3]
                         const float* __restrict__ boff,   // [3]
                         float* __restrict__ offs,         // [M, 3]
                         int M) {
    __shared__ __align__(32) _Float16 Bsh[16 * C_DIM];   // [col][k], 8 KB

    const int lane = threadIdx.x & 31;
    const int tile = blockIdx.x;
    const int sub  = lane & 15;
    const bool hi  = lane >= 16;

    // ---- stage B: zero-fill 16x256 halfs, then write cols 0..2 transposed ----
    {
        uint32_t* bz = (uint32_t*)Bsh;                   // 2048 dwords
        #pragma unroll
        for (int i = 0; i < 2048 / 32; ++i) bz[lane + i * 32] = 0u;
        __syncthreads();
        #pragma unroll
        for (int i = 0; i < 768 / 32; ++i) {
            int idx = lane + i * 32;                     // idx = k*3 + col
            int k = idx / 3, col = idx - 3 * k;
            Bsh[col * C_DIM + k] = (_Float16)Woff[idx];
        }
        __syncthreads();
    }

    const int row = tile * 16 + sub;
    const int rowClamped = row < M ? row : (M - 1);
    const float* arow = qc + (size_t)rowClamped * C_DIM;

    const int kaBase0 = hi ? 16 : 0;
    const int kaBase1 = hi ? 24 : 8;
    const int kbBase  = hi ? 16 : 0;
    const _Float16* bcol = &Bsh[sub * C_DIM + kbBase];   // 32B-aligned

    v8f acc = {};
    #pragma unroll
    for (int k0 = 0; k0 < C_DIM; k0 += 32) {
        v16h a;
        #pragma unroll
        for (int j = 0; j < 8; ++j)  a[j]     = (_Float16)arow[k0 + kaBase0 + j];
        #pragma unroll
        for (int j = 0; j < 8; ++j)  a[8 + j] = (_Float16)arow[k0 + kaBase1 + j];

        v16h b = *(const v16h*)(bcol + k0);              // 2x ds_load_b128

        acc = __builtin_amdgcn_wmma_f32_16x16x32_f16(
            /*neg_a=*/false, a, /*neg_b=*/false, b,
            /*c_mod=*/(short)0, acc, /*reuse_a=*/false, /*reuse_b=*/false);
    }

    if (sub < 3) {
        const float bias = boff[sub];
        const int mbase = tile * 16 + (hi ? 8 : 0);
        #pragma unroll
        for (int r = 0; r < 8; ++r) {
            int m = mbase + r;
            if (m < M) offs[(size_t)m * 3 + sub] = acc[r] + bias;
        }
    }
}

// -----------------------------------------------------------------------------
// Kernel 2: per-point multi-level bilinear sample + softmax(z-gaussian) blend,
// broadcast to P_IN copies. One block per (b,n), one thread per channel.
// Broadcast store done via LDS transpose so stores are global_store_b128.
// -----------------------------------------------------------------------------
__global__ void __launch_bounds__(C_DIM)
sample3d_kernel(const float* __restrict__ f0, const float* __restrict__ f1,
                const float* __restrict__ f2, const float* __restrict__ f3,
                const float* __restrict__ qpos,    // [B,N,4]
                const float* __restrict__ offs,    // [B*N,3]
                const float* __restrict__ sigma,   // [1]
                float* __restrict__ out,           // [B,N,P_IN,C]
                int M) {
    __shared__ __align__(16) float sh[C_DIM];

    const int c  = threadIdx.x;          // channel 0..255
    const int pt = blockIdx.x;           // 0..B*N-1
    const int b  = pt / N_QUERIES;

    // per-point scalars (uniform across lanes -> broadcast)
    const float x = qpos[(size_t)pt * 4 + 0];
    const float y = qpos[(size_t)pt * 4 + 1];
    const float z = qpos[(size_t)pt * 4 + 2];
    const float r = qpos[(size_t)pt * 4 + 3];
    const float dx = offs[(size_t)pt * 3 + 0];
    const float dy = offs[(size_t)pt * 3 + 1];
    const float dz = offs[(size_t)pt * 3 + 2];

    const float sx = x + dx * exp2f(z - r);
    const float sy = y + dy * exp2f(z + r);
    const float sz = z + dz;

    const float sg = sigma[0];
    const float inv2s2 = 1.0f / (2.0f * sg * sg);

    // reference: w_l = exp(-(sz-l)^2/(2s^2)); zw = softmax(w)  (softmax applied
    // to the already-exponentiated values). w_l in (0,1] -> numerically safe.
    float wnum[N_LEVELS];
    float zsum = 0.0f;
    #pragma unroll
    for (int l = 0; l < N_LEVELS; ++l) {
        float d = sz - (float)l;
        float g = __expf(-d * d * inv2s2);
        wnum[l] = __expf(g);
        zsum += wnum[l];
    }
    const float inv_zsum = 1.0f / zsum;

    const float* feats[N_LEVELS] = {f0, f1, f2, f3};
    const int    DIM[N_LEVELS]   = {100, 50, 25, 13};

    float acc = 0.0f;
    #pragma unroll
    for (int l = 0; l < N_LEVELS; ++l) {
        const int H = DIM[l], W = DIM[l];
        // border padding, align_corners=False: px = clamp(sx-0.5, 0, W-1)
        float px = fminf(fmaxf(sx - 0.5f, 0.0f), (float)(W - 1));
        float py = fminf(fmaxf(sy - 0.5f, 0.0f), (float)(H - 1));
        float x0f = floorf(px), y0f = floorf(py);
        float wx = px - x0f,    wy = py - y0f;
        int x0 = (int)x0f,      y0 = (int)y0f;
        int x1 = min(x0 + 1, W - 1);
        int y1 = min(y0 + 1, H - 1);

        const float* base = feats[l] + ((size_t)b * C_DIM + c) * (size_t)(H * W);
        float v00 = base[y0 * W + x0];
        float v01 = base[y0 * W + x1];
        float v10 = base[y1 * W + x0];
        float v11 = base[y1 * W + x1];

        float bil = v00 * (1.0f - wx) * (1.0f - wy)
                  + v01 * wx          * (1.0f - wy)
                  + v10 * (1.0f - wx) * wy
                  + v11 * wx          * wy;
        acc += (wnum[l] * inv_zsum) * bil;
    }

    // ---- broadcast to P_IN copies with b128 stores ----
    sh[c] = acc;
    __syncthreads();

    // flat output f = pass*1024 + tid*4 ; channel = f mod 256 = (tid*4) & 255
    const float4 v4 = *(const float4*)&sh[(threadIdx.x * 4) & (C_DIM - 1)];
    float* obase = out + (size_t)pt * (P_IN * C_DIM);
    #pragma unroll
    for (int pass = 0; pass < (P_IN * C_DIM) / (C_DIM * 4); ++pass) {   // 8
        *(float4*)(obase + pass * (C_DIM * 4) + threadIdx.x * 4) = v4;
    }
}

extern "C" void kernel_launch(void* const* d_in, const int* in_sizes, int n_in,
                              void* d_out, int out_size, void* d_ws, size_t ws_size,
                              hipStream_t stream) {
    const float* feat0 = (const float*)d_in[0];
    const float* feat1 = (const float*)d_in[1];
    const float* feat2 = (const float*)d_in[2];
    const float* feat3 = (const float*)d_in[3];
    const float* qpos  = (const float*)d_in[4];
    const float* qcont = (const float*)d_in[5];
    const float* Woff  = (const float*)d_in[6];
    const float* boff  = (const float*)d_in[7];
    const float* sigma = (const float*)d_in[8];

    const int M = in_sizes[4] / 4;          // B*N (2400)
    float* offs = (float*)d_ws;             // [M,3] scratch, 28.8 KB

    // 1) offsets GEMM via WMMA (one wave per 16-row tile)
    const int tiles = (M + 15) / 16;
    offsets_gemm_wmma_kernel<<<tiles, 32, 0, stream>>>(qcont, Woff, boff, offs, M);

    // 2) multi-level bilinear sample + blend + broadcast
    sample3d_kernel<<<M, C_DIM, 0, stream>>>(feat0, feat1, feat2, feat3,
                                             qpos, offs, sigma,
                                             (float*)d_out, M);
}
